// ODELSTM_40458591928887
// MI455X (gfx1250) — matrix operand order repaired and checked by reading, not compiled, not run on hardware
//
#include <hip/hip_runtime.h>
#include <stdint.h>
#include <stddef.h>

// Problem dims
#define B_N   128
#define S_N   512
#define H_N   512
#define OH_N  256
#define G4H   (4 * H_N)
#define L_N   2

// Grid: 2 layer-pipelines x 8 batch groups (16 rows) x 16 N-slices = 256 WGs
#define NSLICE 16
#define NGROUP 8
#define NWGT   (L_N * NGROUP * NSLICE)
#define TPB    256   // 8 waves of 32
#define R_DEPTH 4    // h0 ring depth (layer0 -> layer1)

typedef __attribute__((ext_vector_type(16))) __bf16 v16bf;
typedef __attribute__((ext_vector_type(8)))  float  v8f;
typedef __attribute__((ext_vector_type(4)))  unsigned int u32x4;
typedef __attribute__((ext_vector_type(2)))  unsigned int u32x2;
typedef __attribute__((ext_vector_type(4)))  float  f32x4;

static constexpr size_t OUTS_ELEMS = (size_t)B_N * S_N * H_N;              // outs (B,S,H)
static constexpr size_t OUT_HT     = OUTS_ELEMS;                           // hT (L,B,H)
static constexpr size_t OUT_CT     = OUTS_ELEMS + (size_t)L_N * B_N * H_N; // cT

// ---------- helpers ----------
__device__ __forceinline__ unsigned short f2bf(float f) {
  unsigned u = __builtin_bit_cast(unsigned, f);
  u += 0x7FFFu + ((u >> 16) & 1u);          // round-to-nearest-even
  return (unsigned short)(u >> 16);
}
__device__ __forceinline__ float sigmoidf_(float x) {
  return 1.0f / (1.0f + __expf(-x));
}

// A-fragment (16x32 bf16), row-major source (global bf16 buffers).
// ISA layout: lanes 0-15: V0-3 = K 0..7, V4-7 = K 16..23; lanes 16-31: +8.
__device__ __forceinline__ v16bf load_a_bf16(const unsigned short* rowbase,
                                             size_t rstride, int kk, int lane) {
  const int m  = lane & 15;
  const int kb = kk + ((lane >> 4) << 3);
  const unsigned short* p = rowbase + (size_t)m * rstride + kb;
  union { v16bf v; u32x4 q[2]; } u;
  u.q[0] = *(const u32x4*)(p);
  u.q[1] = *(const u32x4*)(p + 16);
  return u.v;
}

// B-fragment (32x16 bf16) for C = A * W^T: B[k][n] = W[n][k], W row-major (n,k).
// Lane n = n0 + (lane&15), k at kk + 8*(lane>=16) + {0..7, 16..23} (ISA A/B order).
// Non-temporal: streamed gate-weight slices exceed WGP$ (no intra-step reuse).
__device__ __forceinline__ v16bf load_b_nt(const unsigned short* base, size_t rstride,
                                           int n0, int kk, int lane) {
  const int n  = n0 + (lane & 15);
  const int kb = kk + ((lane >> 4) << 3);
  const unsigned short* p = base + (size_t)n * rstride + kb;
  union { v16bf v; u32x4 q[2]; } u;
  u.q[0] = __builtin_nontemporal_load((const u32x4*)(p));
  u.q[1] = __builtin_nontemporal_load((const u32x4*)(p + 16));
  return u.v;
}

// Same fragments from LDS-resident tiles (inlined => addrspace(3) ds loads).
__device__ __forceinline__ v16bf load_a_lds(const unsigned short* rowbase,
                                            int rstride, int kk, int lane) {
  const int m  = lane & 15;
  const int kb = kk + ((lane >> 4) << 3);
  const unsigned short* p = rowbase + m * rstride + kb;
  union { v16bf v; u32x4 q[2]; } u;
  u.q[0] = *(const u32x4*)(p);
  u.q[1] = *(const u32x4*)(p + 16);
  return u.v;
}
__device__ __forceinline__ v16bf load_b_lds(const unsigned short* base, int rstride,
                                            int n0, int kk, int lane) {
  const int n  = n0 + (lane & 15);
  const int kb = kk + ((lane >> 4) << 3);
  const unsigned short* p = base + n * rstride + kb;
  union { v16bf v; u32x4 q[2]; } u;
  u.q[0] = *(const u32x4*)(p);
  u.q[1] = *(const u32x4*)(p + 16);
  return u.v;
}

__device__ __forceinline__ v8f wmma_bf16(v16bf a, v16bf b, v8f c) {
  return __builtin_amdgcn_wmma_f32_16x16x32_bf16(false, a, false, b,
                                                 (short)0, c, false, false);
}

// Group barrier: 16 WGs, monotone counter (re-zeroed by init kernel each call).
__device__ __forceinline__ void group_barrier(unsigned* ctr, unsigned& epoch) {
  epoch++;
  __builtin_amdgcn_fence(__ATOMIC_RELEASE, "agent");
  __syncthreads();
  if (threadIdx.x == 0) {
    __hip_atomic_fetch_add(ctr, 1u, __ATOMIC_RELEASE, __HIP_MEMORY_SCOPE_AGENT);
    const unsigned target = epoch * NSLICE;
    while (__hip_atomic_load(ctr, __ATOMIC_RELAXED, __HIP_MEMORY_SCOPE_AGENT) < target) {
      __builtin_amdgcn_s_sleep(1);
    }
  }
  __syncthreads();
  __builtin_amdgcn_fence(__ATOMIC_ACQUIRE, "agent");
}

// Wait until *ctr >= target (cross-pipeline producer/consumer flag).
__device__ __forceinline__ void wait_ctr_ge(unsigned* ctr, unsigned target) {
  __syncthreads();
  if (threadIdx.x == 0) {
    while (__hip_atomic_load(ctr, __ATOMIC_RELAXED, __HIP_MEMORY_SCOPE_AGENT) < target) {
      __builtin_amdgcn_s_sleep(1);
    }
  }
  __syncthreads();
  __builtin_amdgcn_fence(__ATOMIC_ACQUIRE, "agent");
}

// ---------- init: fp32 -> bf16 weight conversion + counter reset ----------
#define NIH_ELEMS ((size_t)L_N * G4H * H_N)   // 2*2048*512
#define NW1_ELEMS ((size_t)L_N * OH_N * H_N)  // 2*256*512
#define NW2_ELEMS ((size_t)L_N * H_N * OH_N)  // 2*512*256
#define CONV_TOTAL (NIH_ELEMS * 2 + NW1_ELEMS + NW2_ELEMS)
#define NCTR (L_N * NGROUP + 2 * NGROUP)      // 16 barriers + 8 prod + 8 cons

__global__ void odelstm_init(const float* __restrict__ Wih, const float* __restrict__ Whh,
                             const float* __restrict__ W1,  const float* __restrict__ W2,
                             unsigned short* __restrict__ oWih, unsigned short* __restrict__ oWhh,
                             unsigned short* __restrict__ oW1,  unsigned short* __restrict__ oW2,
                             unsigned* __restrict__ ctrs) {
  if (blockIdx.x == 0 && threadIdx.x < NCTR) ctrs[threadIdx.x] = 0u;
  size_t i = (size_t)blockIdx.x * TPB + threadIdx.x;
  if (i < NIH_ELEMS) {
    oWih[i] = f2bf(Wih[i]);
  } else if (i < 2 * NIH_ELEMS) {
    size_t j = i - NIH_ELEMS;                 oWhh[j] = f2bf(Whh[j]);
  } else if (i < 2 * NIH_ELEMS + NW1_ELEMS) {
    size_t j = i - 2 * NIH_ELEMS;             oW1[j]  = f2bf(W1[j]);
  } else if (i < CONV_TOTAL) {
    size_t j = i - 2 * NIH_ELEMS - NW1_ELEMS; oW2[j]  = f2bf(W2[j]);
  }
}

// ---------- persistent ODE-LSTM kernel (layer-pipelined over time) ----------
__global__ __launch_bounds__(TPB) void odelstm_persistent(
    const float* __restrict__ inputs,     // (B,S,H)
    const float* __restrict__ timespans,  // (B,S)
    const float* __restrict__ b_ih,       // (L,4H)
    const float* __restrict__ b_hh,       // (L,4H)
    const float* __restrict__ b1,         // (L,OH)
    const float* __restrict__ b2,         // (L,H)
    const unsigned short* __restrict__ Wih,  // (L,4H,H) bf16
    const unsigned short* __restrict__ Whh,  // (L,4H,H) bf16
    const unsigned short* __restrict__ W1,   // (L,OH,H) bf16
    const unsigned short* __restrict__ W2,   // (L,H,OH) bf16
    unsigned short* __restrict__ hbf,        // (L,B,H)   bf16 recurrent state
    unsigned short* __restrict__ ybf,        // (L,B,H)   bf16 RK4 working vec
    unsigned short* __restrict__ zbf,        // (L,B,OH)  bf16 tanh(.W1^T+b1)
    unsigned short* __restrict__ ring,       // (R,B,H)   bf16 h0[t] -> layer1
    unsigned* __restrict__ ctrs,             // barriers | prod | cons
    float* __restrict__ out)                 // outs | hT | cT
{
  const int wg   = blockIdx.x;
  const int lay  = wg >> 7;              // which layer pipeline
  const int r    = wg & 127;
  const int g    = r >> 4;               // batch group: rows m0..m0+15
  const int s    = r & 15;               // N-slice
  const int m0   = g * 16;
  const int tid  = threadIdx.x;
  const int wave = tid >> 5;
  const int lane = tid & 31;
  unsigned* gctr    = ctrs + (lay * NGROUP + g);
  unsigned* prodctr = ctrs + (L_N * NGROUP) + g;
  unsigned* consctr = ctrs + (L_N * NGROUP) + NGROUP + g;
  unsigned epoch = 0;

  __shared__ float stage[8][16][16];                       // GEMM tiles / partials
  __shared__ float c_loc[16][32];                          // persistent cell state
  __shared__ float hnew_loc[16][32];                       // LSTM output (pre-ODE)
  __shared__ float acc_loc[16][32];                        // RK4 k-accumulator
  __shared__ float dt_loc[16];
  __shared__ __align__(16) unsigned short xlds[16][H_N];   // 16KB x_t tile (layer0)
  __shared__ __align__(16) unsigned short w1lds[16 * H_N]; // 16KB W1 slice (own layer)
  __shared__ __align__(16) unsigned short w2lds[32 * OH_N];// 16KB W2 slice (own layer)

  // ---- one-time: zero states, preload LDS-resident fode weight slices ----
  for (int e = tid; e < 16 * 32; e += TPB) ((float*)c_loc)[e] = 0.0f;
  for (int e = tid; e < 16 * 32; e += TPB) {
    int m = e >> 5, hc = 32 * s + (e & 31);
    hbf[((size_t)lay * B_N + m0 + m) * H_N + hc] = 0;      // bf16(0)
  }
  {
    const u32x4* src = (const u32x4*)(W1 + ((size_t)lay * OH_N + 16 * s) * H_N);
    u32x4* dst = (u32x4*)&w1lds[0];
    for (int e = tid; e < (16 * H_N) / 8; e += TPB) dst[e] = src[e];
  }
  {
    const u32x4* src = (const u32x4*)(W2 + ((size_t)lay * H_N + 32 * s) * OH_N);
    u32x4* dst = (u32x4*)&w2lds[0];
    for (int e = tid; e < (32 * OH_N) / 8; e += TPB) dst[e] = src[e];
  }
  group_barrier(gctr, epoch);

  const float RKA[3] = {0.5f, 0.5f, 1.0f};
  const float RKW[3] = {1.0f, 2.0f, 2.0f};

  const unsigned short* Wihl = Wih + (size_t)lay * G4H * H_N;
  const unsigned short* Whhl = Whh + (size_t)lay * G4H * H_N;
  unsigned short* hb = hbf + (size_t)lay * B_N * H_N;
  unsigned short* yb = ybf + (size_t)lay * B_N * H_N;
  unsigned short* zb = zbf + (size_t)lay * B_N * OH_N;

  for (int t = 0; t < S_N; ++t) {
    // cross-pipeline flow control
    if (lay == 1)            wait_ctr_ge(prodctr, (unsigned)(t + 1));       // h0[t] ready
    else if (t >= R_DEPTH)   wait_ctr_ge(consctr, (unsigned)(t - R_DEPTH + 1)); // slot free

    // stage dt (+ x_t tile fp32->bf16 for layer0, once, vectorized)
    if (tid < 16) dt_loc[tid] = timespans[(size_t)(m0 + tid) * S_N + t];
    if (lay == 0) {
      const float* xsrc = inputs + ((size_t)m0 * S_N + t) * H_N;
      for (int e = tid; e < (16 * H_N) / 4; e += TPB) {
        int m  = e >> 7;                 // 128 float4 per row
        int k4 = (e & 127) << 2;
        f32x4 f = *(const f32x4*)(xsrc + (size_t)m * (S_N * H_N) + k4);
        u32x2 pk;
        pk.x = (unsigned)f2bf(f.x) | ((unsigned)f2bf(f.y) << 16);
        pk.y = (unsigned)f2bf(f.z) | ((unsigned)f2bf(f.w) << 16);
        *(u32x2*)&xlds[m][k4] = pk;
      }
    }
    __syncthreads();

    // ---- fused gates GEMM: gates = inp@Wih^T + h@Whh^T (bias in pointwise) ----
    {
      const int ht = wave >> 2;            // which H-16-tile of the 32-col slice
      const int gt = wave & 3;             // gate index i/f/g/o
      const int n0 = gt * H_N + 32 * s + 16 * ht;
      v8f c = {0.f, 0.f, 0.f, 0.f, 0.f, 0.f, 0.f, 0.f};
      if (lay == 0) {
        for (int kk = 0; kk < H_N; kk += 32) {
          v16bf a = load_a_lds(&xlds[0][0], H_N, kk, lane);
          v16bf b = load_b_nt(Wihl, H_N, n0, kk, lane);
          c = wmma_bf16(a, b, c);
        }
      } else {
        const unsigned short* A0 = ring + ((size_t)(t % R_DEPTH) * B_N + m0) * H_N;
        for (int kk = 0; kk < H_N; kk += 32) {
          v16bf a = load_a_bf16(A0, H_N, kk, lane);
          v16bf b = load_b_nt(Wihl, H_N, n0, kk, lane);
          c = wmma_bf16(a, b, c);
        }
      }
      {
        const unsigned short* Ah = hb + (size_t)m0 * H_N;
        for (int kk = 0; kk < H_N; kk += 32) {
          v16bf a = load_a_bf16(Ah, H_N, kk, lane);
          v16bf b = load_b_nt(Whhl, H_N, n0, kk, lane);
          c = wmma_bf16(a, b, c);
        }
      }
#pragma unroll
      for (int v = 0; v < 8; ++v)
        stage[wave][v + ((lane >> 4) << 3)][lane & 15] = c[v];
    }
    __syncthreads();

    // ---- LSTM pointwise (512 elems: 16 rows x 32 cols of this slice) ----
    for (int e = tid; e < 512; e += TPB) {
      const int m = e >> 5, hcl = e & 31;
      const int ht = hcl >> 4, nl = hcl & 15;
      const int hc = 32 * s + hcl;
      const size_t bo = (size_t)lay * G4H + hc;
      float gi = stage[ht * 4 + 0][m][nl] + b_ih[bo + 0 * H_N] + b_hh[bo + 0 * H_N];
      float gf = stage[ht * 4 + 1][m][nl] + b_ih[bo + 1 * H_N] + b_hh[bo + 1 * H_N];
      float gg = stage[ht * 4 + 2][m][nl] + b_ih[bo + 2 * H_N] + b_hh[bo + 2 * H_N];
      float go = stage[ht * 4 + 3][m][nl] + b_ih[bo + 3 * H_N] + b_hh[bo + 3 * H_N];
      float i_ = sigmoidf_(gi), f_ = sigmoidf_(gf);
      float g_ = tanhf(gg),     o_ = sigmoidf_(go);
      float cn = f_ * c_loc[m][hcl] + i_ * g_;
      float hn = o_ * tanhf(cn);
      c_loc[m][hcl] = cn;
      hnew_loc[m][hcl] = hn;
      acc_loc[m][hcl]  = 0.0f;
      yb[(size_t)(m0 + m) * H_N + hc] = f2bf(hn);          // y0 = h_new for k1
      if (t == S_N - 1)
        out[OUT_CT + ((size_t)lay * B_N + m0 + m) * H_N + hc] = cn;
    }
    group_barrier(gctr, epoch);   // full y rows visible within the group
    if (lay == 1 && s == 0 && tid == 0)       // ring slot consumed in gates above
      __hip_atomic_fetch_add(consctr, 1u, __ATOMIC_RELEASE, __HIP_MEMORY_SCOPE_AGENT);

    // ---- RK4: 4 x (z = tanh(y@W1^T+b1); k = z@W2^T+b2; update) ----
    for (int j = 0; j < 4; ++j) {
      // GEMM1: one 16-col z tile, K=512 split over waves 0..3 (LDS weights)
      if (wave < 4) {
        const unsigned short* A0 = yb + (size_t)m0 * H_N;
        v8f c = {0.f, 0.f, 0.f, 0.f, 0.f, 0.f, 0.f, 0.f};
        const int k0 = 128 * wave;
        for (int kk = k0; kk < k0 + 128; kk += 32)
          c = wmma_bf16(load_a_bf16(A0, H_N, kk, lane),
                        load_b_lds(&w1lds[0], H_N, 0, kk, lane), c);
#pragma unroll
        for (int v = 0; v < 8; ++v)
          stage[wave][v + ((lane >> 4) << 3)][lane & 15] = c[v];
      }
      __syncthreads();
      {                 // reduce partials + bias + tanh -> z (one elem/thread)
        const int m = tid >> 4, nzl = tid & 15;
        float zv = stage[0][m][nzl] + stage[1][m][nzl] +
                   stage[2][m][nzl] + stage[3][m][nzl] +
                   b1[(size_t)lay * OH_N + 16 * s + nzl];
        zv = tanhf(zv);
        zb[(size_t)(m0 + m) * OH_N + 16 * s + nzl] = f2bf(zv);
      }
      group_barrier(gctr, epoch); // full z visible

      // GEMM2: two 16-col H tiles, K=256 split over 8 waves (LDS weights)
      {
        const int tw = wave & 1;        // which H tile
        const int kc = wave >> 1;       // K chunk of 64
        const unsigned short* A0 = zb + (size_t)m0 * OH_N;
        v8f c = {0.f, 0.f, 0.f, 0.f, 0.f, 0.f, 0.f, 0.f};
        const int k0 = kc * 64;
        for (int kk = k0; kk < k0 + 64; kk += 32)
          c = wmma_bf16(load_a_bf16(A0, OH_N, kk, lane),
                        load_b_lds(&w2lds[0], OH_N, 16 * tw, kk, lane), c);
#pragma unroll
        for (int v = 0; v < 8; ++v)
          stage[wave][v + ((lane >> 4) << 3)][lane & 15] = c[v];
      }
      __syncthreads();
      // reduce partials + RK4 pointwise over all 512 slice elems
      for (int e = tid; e < 512; e += TPB) {
        const int m = e >> 5, hcl = e & 31;
        const int tw = hcl >> 4, nl = hcl & 15;
        const int n = 32 * s + hcl;
        float kv = stage[0 * 2 + tw][m][nl] + stage[1 * 2 + tw][m][nl] +
                   stage[2 * 2 + tw][m][nl] + stage[3 * 2 + tw][m][nl] +
                   b2[(size_t)lay * H_N + n];
        const float dtm = dt_loc[m];
        if (j < 3) {
          acc_loc[m][hcl] += RKW[j] * kv;
          float yv = hnew_loc[m][hcl] + RKA[j] * dtm * kv;
          yb[(size_t)(m0 + m) * H_N + n] = f2bf(yv);
        } else {
          float hf = hnew_loc[m][hcl] + (dtm * (1.0f / 6.0f)) * (acc_loc[m][hcl] + kv);
          unsigned short hfb = f2bf(hf);
          hb[(size_t)(m0 + m) * H_N + n] = hfb;
          if (lay == 0)   // publish h0[t] to layer-1 pipeline
            ring[((size_t)(t % R_DEPTH) * B_N + m0 + m) * H_N + n] = hfb;
          else
            out[((size_t)(m0 + m) * S_N + t) * H_N + n] = hf;
          if (t == S_N - 1)
            out[OUT_HT + ((size_t)lay * B_N + m0 + m) * H_N + n] = hf;
        }
      }
      group_barrier(gctr, epoch); // next y (or h/ring) visible
    } // j

    if (lay == 0 && s == 0 && tid == 0)       // h0[t] fully published
      __hip_atomic_fetch_add(prodctr, 1u, __ATOMIC_RELEASE, __HIP_MEMORY_SCOPE_AGENT);
  } // t
}

// ---------- host ----------
extern "C" void kernel_launch(void* const* d_in, const int* in_sizes, int n_in,
                              void* d_out, int out_size, void* d_ws, size_t ws_size,
                              hipStream_t stream) {
  (void)in_sizes; (void)n_in; (void)out_size; (void)ws_size;
  const float* inputs    = (const float*)d_in[0];
  const float* timespans = (const float*)d_in[1];
  const float* W_ih      = (const float*)d_in[2];
  const float* W_hh      = (const float*)d_in[3];
  const float* b_ih      = (const float*)d_in[4];
  const float* b_hh      = (const float*)d_in[5];
  const float* W1        = (const float*)d_in[6];
  const float* b1        = (const float*)d_in[7];
  const float* W2        = (const float*)d_in[8];
  const float* b2        = (const float*)d_in[9];

  char* ws = (char*)d_ws;
  size_t off = 0;
  unsigned* ctrs = (unsigned*)(ws + off);              off += 256;
  unsigned short* oWih = (unsigned short*)(ws + off);  off += NIH_ELEMS * 2;
  unsigned short* oWhh = (unsigned short*)(ws + off);  off += NIH_ELEMS * 2;
  unsigned short* oW1  = (unsigned short*)(ws + off);  off += NW1_ELEMS * 2;
  unsigned short* oW2  = (unsigned short*)(ws + off);  off += NW2_ELEMS * 2;
  unsigned short* hbf  = (unsigned short*)(ws + off);  off += (size_t)L_N * B_N * H_N * 2;
  unsigned short* ybf  = (unsigned short*)(ws + off);  off += (size_t)L_N * B_N * H_N * 2;
  unsigned short* zbf  = (unsigned short*)(ws + off);  off += (size_t)L_N * B_N * OH_N * 2;
  unsigned short* ring = (unsigned short*)(ws + off);  off += (size_t)R_DEPTH * B_N * H_N * 2;

  const unsigned convBlocks = (unsigned)((CONV_TOTAL + TPB - 1) / TPB);
  odelstm_init<<<convBlocks, TPB, 0, stream>>>(W_ih, W_hh, W1, W2,
                                               oWih, oWhh, oW1, oW2, ctrs);
  odelstm_persistent<<<NWGT, TPB, 0, stream>>>(inputs, timespans, b_ih, b_hh, b1, b2,
                                               oWih, oWhh, oW1, oW2,
                                               hbf, ybf, zbf, ring, ctrs, (float*)d_out);
}
